// AttentionHead_61022895341633
// MI455X (gfx1250) — matrix operand order repaired and checked by reading, not compiled, not run on hardware
//
#include <hip/hip_runtime.h>
#include <hip/hip_bf16.h>
#include <stdint.h>

// ---------------------------------------------------------------------------
// MI455X (gfx1250, wave32) fused multi-head attention + head-averaged pooling.
// All matmuls run through v_wmma_f32_16x16x32_bf16. attn [B,H,T,T] is never
// materialized (flash-style two-pass per 16-row Q tile); column sums of the
// softmax (needed for pool_w) are accumulated with global f32 atomics.
// GEMM tile staging uses CDNA5 async global->LDS DMA (ASYNCcnt path) with
// double buffering: tile i+1's DMA overlaps tile i's WMMA work
// (async loads complete in order, so s_wait_asynccnt 4 gates only tile i).
// ---------------------------------------------------------------------------

typedef __bf16 bf16_t;
typedef __bf16        v16bf __attribute__((ext_vector_type(16)));
typedef float         v8f   __attribute__((ext_vector_type(8)));
typedef float         f32x4 __attribute__((ext_vector_type(4)));
typedef unsigned int  u32x4 __attribute__((ext_vector_type(4)));

#define WMMA_BF16(A, B, C) \
  __builtin_amdgcn_wmma_f32_16x16x32_bf16(false, (A), false, (B), (short)0, (C), false, false)

union ABu { v16bf v; u32x4 u[2]; };

// A-operand (16x32 bf16): lane m = lane%16 row, K 0-15 lanes<16,
// K 16-31 lanes>=16. base = row start (32 contiguous K elems, 16B aligned).
__device__ __forceinline__ v16bf load_a32(const bf16_t* base, int h2) {
  ABu r;
  const u32x4* p = (const u32x4*)base;
  r.u[0] = p[2 * h2];
  r.u[1] = p[2 * h2 + 1];
  return r.v;
}
// B-operand (32x16 bf16): lane n = lane%16 column, K 0-15 for lanes<16,
// K 16-31 for lanes>=16. base = column start (32 contiguous K elems).
__device__ __forceinline__ v16bf load_b32(const bf16_t* base, int h2) {
  ABu r;
  const u32x4* p = (const u32x4*)base;
  r.u[0] = p[2 * h2];
  r.u[1] = p[2 * h2 + 1];
  return r.v;
}

// 32-byte async global->LDS copy for one lane (INST_OFFSET applies to both
// the global and LDS side, CDNA5 ISA). Tracked by ASYNCcnt.
__device__ __forceinline__ void async_copy32(unsigned lds_addr, const bf16_t* gptr) {
  unsigned long long ga = (unsigned long long)(size_t)gptr;
  asm volatile(
      "global_load_async_to_lds_b128 %0, %1, off\n\t"
      "global_load_async_to_lds_b128 %0, %1, off offset:16"
      :: "v"(lds_addr), "v"(ga) : "memory");
}
__device__ __forceinline__ void wait_async0() {
  asm volatile("s_wait_asynccnt 0x0" ::: "memory");
}
// allow 4 async ops (the just-issued next tile) to remain in flight
__device__ __forceinline__ void wait_async4() {
  asm volatile("s_wait_asynccnt 0x4" ::: "memory");
}

__device__ __forceinline__ float redmax16(float v) {
#pragma unroll
  for (int m = 1; m < 16; m <<= 1) v = fmaxf(v, __shfl_xor(v, m, 32));
  return v;
}
__device__ __forceinline__ float redsum16(float v) {
#pragma unroll
  for (int m = 1; m < 16; m <<= 1) v += __shfl_xor(v, m, 32);
  return v;
}

// --------------------------- conversion kernels ----------------------------

__global__ void k_cvt_x(const float* __restrict__ x, bf16_t* __restrict__ xb, int n8) {
  int i = blockIdx.x * blockDim.x + threadIdx.x;
  if (i >= n8) return;
  const f32x4* p = (const f32x4*)x + (size_t)i * 2;
  f32x4 a = p[0], b = p[1];
  union { bf16_t h[8]; u32x4 u; } o;
#pragma unroll
  for (int j = 0; j < 4; ++j) { o.h[j] = (bf16_t)a[j]; o.h[4 + j] = (bf16_t)b[j]; }
  ((u32x4*)xb)[i] = o.u;
}

// W [D,D] fp32 -> Wt [D_out][D_in] bf16 (transposed so B-operand K is contiguous)
__global__ void k_wT(const float* __restrict__ w, bf16_t* __restrict__ wt) {
  int i = blockIdx.x * blockDim.x + threadIdx.x;   // 0 .. 1M-1
  int r = i >> 10, c = i & 1023;
  wt[(size_t)c * 1024 + r] = (bf16_t)w[i];
}

// ------------------------------ QKV GEMM -----------------------------------
// grid (D/64, BT/64, 3); block 128 (4 waves). Each wave owns a 16x64 tile.
// mode 0: Q (x0.125) -> [B,H,T,HD]; mode 1: K -> [B,H,T,HD]; mode 2: V -> [B,H,HD,T]
// K-loop staging: double-buffered async global->LDS DMA overlapped with WMMA.
__global__ void k_qkv(const bf16_t* __restrict__ xb,
                      const bf16_t* __restrict__ Wqt, const bf16_t* __restrict__ Wkt,
                      const bf16_t* __restrict__ Wvt,
                      const float* __restrict__ bq, const float* __restrict__ bk,
                      const float* __restrict__ bv,
                      bf16_t* __restrict__ Qb, bf16_t* __restrict__ Kb,
                      bf16_t* __restrict__ Vtb) {
  __shared__ __align__(16) bf16_t sx[2][64][32];
  __shared__ __align__(16) bf16_t sw[2][64][32];
  const int mode = blockIdx.z;
  const bf16_t* Wt  = (mode == 0) ? Wqt : ((mode == 1) ? Wkt : Wvt);
  const float*  bia = (mode == 0) ? bq  : ((mode == 1) ? bk  : bv);
  const int tid = threadIdx.x;
  const int wave = tid >> 5, lane = tid & 31, ln = lane & 15, h2 = lane >> 4;
  const int m0 = blockIdx.y * 64, n0 = blockIdx.x * 64;
  const int srow = tid >> 1, shalf = tid & 1;

  unsigned lx[2], lw[2];
  lx[0] = (unsigned)(size_t)&sx[0][srow][shalf * 16];
  lx[1] = (unsigned)(size_t)&sx[1][srow][shalf * 16];
  lw[0] = (unsigned)(size_t)&sw[0][srow][shalf * 16];
  lw[1] = (unsigned)(size_t)&sw[1][srow][shalf * 16];
  const bf16_t* gxrow = xb + (size_t)(m0 + srow) * 1024 + shalf * 16;
  const bf16_t* gwrow = Wt + (size_t)(n0 + srow) * 1024 + shalf * 16;

  // prologue: DMA first K-tile into buffer 0
  async_copy32(lx[0], gxrow);
  async_copy32(lw[0], gwrow);

  v8f acc0 = {}, acc1 = {}, acc2 = {}, acc3 = {};
  for (int k0 = 0; k0 < 1024; k0 += 32) {
    const int buf = (k0 >> 5) & 1;
    if (k0 + 32 < 1024) {
      // issue next tile's DMA into the other buffer, then gate on this tile
      async_copy32(lx[buf ^ 1], gxrow + k0 + 32);
      async_copy32(lw[buf ^ 1], gwrow + k0 + 32);
      wait_async4();
    } else {
      wait_async0();
    }
    __syncthreads();
    v16bf a = load_a32(&sx[buf][wave * 16 + ln][0], h2);
    acc0 = WMMA_BF16(a, load_b32(&sw[buf][ 0 + ln][0], h2), acc0);
    acc1 = WMMA_BF16(a, load_b32(&sw[buf][16 + ln][0], h2), acc1);
    acc2 = WMMA_BF16(a, load_b32(&sw[buf][32 + ln][0], h2), acc2);
    acc3 = WMMA_BF16(a, load_b32(&sw[buf][48 + ln][0], h2), acc3);
    __syncthreads();
  }
  const float scale = (mode == 0) ? 0.125f : 1.0f;   // fold 1/sqrt(HD) into Q
  v8f accs[4] = {acc0, acc1, acc2, acc3};
#pragma unroll
  for (int nt = 0; nt < 4; ++nt) {
    int c = n0 + nt * 16 + ln;
    float bval = bia[c];
    int hh = c >> 6, d = c & 63;
#pragma unroll
    for (int j = 0; j < 8; ++j) {
      int r = m0 + wave * 16 + 8 * h2 + j;
      int bidx = r >> 11, t = r & 2047;
      float v = (accs[nt][j] + bval) * scale;
      if (mode == 2) {
        Vtb[((size_t)(bidx * 16 + hh) * 64 + d) * 2048 + t] = (bf16_t)v;
      } else {
        bf16_t* dst = (mode == 0) ? Qb : Kb;
        dst[((size_t)(bidx * 16 + hh) * 2048 + t) * 64 + d] = (bf16_t)v;
      }
    }
  }
}

// ------------------------- flash attention (2-pass) ------------------------
// grid (T/64, B*H); block 128 (4 waves); wave handles one 16-row Q tile.
__global__ void k_attn(const bf16_t* __restrict__ Qb, const bf16_t* __restrict__ Kb,
                       const bf16_t* __restrict__ Vtb, bf16_t* __restrict__ AO,
                       float* __restrict__ colsum) {
  __shared__ __align__(16) bf16_t sp[4][16][32];   // per-wave P tile (16x32 bf16)
  const int tid = threadIdx.x;
  const int wave = tid >> 5, lane = tid & 31, ln = lane & 15, h2 = lane >> 4;
  const int bh = blockIdx.y, b = bh >> 4, hh = bh & 15;
  const int q0 = blockIdx.x * 64 + wave * 16;

  const bf16_t* qrow = Qb + ((size_t)bh * 2048 + q0 + ln) * 64;
  v16bf qa0 = load_a32(qrow, h2);
  v16bf qa1 = load_a32(qrow + 32, h2);

  float m[8], l[8];
#pragma unroll
  for (int j = 0; j < 8; ++j) { m[j] = -1e30f; l[j] = 0.f; }

  // ---- pass 1: running rowmax / rowsumexp (no O accumulator to rescale) ----
  for (int kt = 0; kt < 2048; kt += 32) {
    const bf16_t* kr0 = Kb + ((size_t)bh * 2048 + kt + ln) * 64;
    const bf16_t* kr1 = kr0 + 16 * 64;
    v8f s1 = {}, s2 = {};
    s1 = WMMA_BF16(qa0, load_b32(kr0,      h2), s1);
    s1 = WMMA_BF16(qa1, load_b32(kr0 + 32, h2), s1);
    s2 = WMMA_BF16(qa0, load_b32(kr1,      h2), s2);
    s2 = WMMA_BF16(qa1, load_b32(kr1 + 32, h2), s2);
#pragma unroll
    for (int j = 0; j < 8; ++j) {
      float a = s1[j], c = s2[j];
      float mn = fmaxf(m[j], redmax16(fmaxf(a, c)));
      float rs = redsum16(__expf(a - mn) + __expf(c - mn));
      l[j] = l[j] * __expf(m[j] - mn) + rs;
      m[j] = mn;
    }
  }
  float rl[8];
#pragma unroll
  for (int j = 0; j < 8; ++j) rl[j] = 1.0f / l[j];

  // ---- pass 2: P = exp(S - m_fin); O += P@V; colsum += colsums(P/l) --------
  v8f o0 = {}, o1 = {}, o2 = {}, o3 = {};
  for (int kt = 0; kt < 2048; kt += 32) {
    const bf16_t* kr0 = Kb + ((size_t)bh * 2048 + kt + ln) * 64;
    const bf16_t* kr1 = kr0 + 16 * 64;
    v8f s1 = {}, s2 = {};
    s1 = WMMA_BF16(qa0, load_b32(kr0,      h2), s1);
    s1 = WMMA_BF16(qa1, load_b32(kr0 + 32, h2), s1);
    s2 = WMMA_BF16(qa0, load_b32(kr1,      h2), s2);
    s2 = WMMA_BF16(qa1, load_b32(kr1 + 32, h2), s2);
    float cs1 = 0.f, cs2 = 0.f;
#pragma unroll
    for (int j = 0; j < 8; ++j) {
      float p1 = __expf(s1[j] - m[j]);
      float p2 = __expf(s2[j] - m[j]);
      sp[wave][8 * h2 + j][ln]      = (bf16_t)p1;   // C-layout -> LDS
      sp[wave][8 * h2 + j][16 + ln] = (bf16_t)p2;
      cs1 += p1 * rl[j];
      cs2 += p2 * rl[j];
    }
    __syncthreads();
    v16bf pa = load_a32(&sp[wave][ln][0], h2);      // LDS -> A-layout
    o0 = WMMA_BF16(pa, load_b32(Vtb + ((size_t)bh * 64 +  0 + ln) * 2048 + kt, h2), o0);
    o1 = WMMA_BF16(pa, load_b32(Vtb + ((size_t)bh * 64 + 16 + ln) * 2048 + kt, h2), o1);
    o2 = WMMA_BF16(pa, load_b32(Vtb + ((size_t)bh * 64 + 32 + ln) * 2048 + kt, h2), o2);
    o3 = WMMA_BF16(pa, load_b32(Vtb + ((size_t)bh * 64 + 48 + ln) * 2048 + kt, h2), o3);
    atomicAdd(&colsum[b * 2048 + kt + ln],       cs1);
    atomicAdd(&colsum[b * 2048 + kt + 16 + ln],  cs2);
    __syncthreads();
  }

  v8f oo[4] = {o0, o1, o2, o3};
#pragma unroll
  for (int nt = 0; nt < 4; ++nt) {
#pragma unroll
    for (int j = 0; j < 8; ++j) {
      int t = q0 + 8 * h2 + j;
      AO[(size_t)(b * 2048 + t) * 1024 + hh * 64 + nt * 16 + ln] =
          (bf16_t)(oo[nt][j] * rl[j]);
    }
  }
}

// ---------------------- pooled reduction (tiny, fp32) ----------------------
// y[b,c] = sum_t pool_w[b,t] * AO[b,t,c];  pool_w = colsum/(H*T)
__global__ void k_pool(const float* __restrict__ colsum, const bf16_t* __restrict__ AO,
                       float* __restrict__ y) {
  int b = blockIdx.y, c = blockIdx.x * 256 + threadIdx.x;
  float acc = 0.f;
  for (int t = 0; t < 2048; ++t)
    acc += colsum[b * 2048 + t] * (float)AO[(size_t)(b * 2048 + t) * 1024 + c];
  y[b * 1024 + c] = acc * (1.0f / (16.0f * 2048.0f));
}

// pooled[b,d] = y[b,:] @ Wo[:,d] + bo[d]   (sum_t pool_w == 1 exactly)
__global__ void k_oproj(const float* __restrict__ y, const float* __restrict__ Wo,
                        const float* __restrict__ bo, float* __restrict__ out) {
  int b = blockIdx.y, d = blockIdx.x * 256 + threadIdx.x;
  float acc = bo[d];
  for (int c = 0; c < 1024; ++c)
    acc += y[b * 1024 + c] * Wo[(size_t)c * 1024 + d];
  out[b * 1024 + d] = acc;
}

// ------------------------------- launcher ----------------------------------

extern "C" void kernel_launch(void* const* d_in, const int* in_sizes, int n_in,
                              void* d_out, int out_size, void* d_ws, size_t ws_size,
                              hipStream_t stream) {
  const float* x  = (const float*)d_in[0];
  const float* Wq = (const float*)d_in[1];
  const float* bq = (const float*)d_in[2];
  const float* Wk = (const float*)d_in[3];
  const float* bk = (const float*)d_in[4];
  const float* Wv = (const float*)d_in[5];
  const float* bv = (const float*)d_in[6];
  const float* Wo = (const float*)d_in[7];
  const float* bo = (const float*)d_in[8];
  float* out = (float*)d_out;

  char* ws = (char*)d_ws;
  const size_t SZ_XB = (size_t)8192 * 1024 * 2;          // 16 MB  x in bf16
  const size_t SZ_WT = (size_t)1024 * 1024 * 2;          //  2 MB  per weight
  const size_t SZ_H  = (size_t)4 * 16 * 2048 * 64 * 2;   // 16 MB  per Q/K/Vt/AO
  bf16_t* xb     = (bf16_t*)(ws);
  bf16_t* Wqt    = (bf16_t*)(ws + SZ_XB);
  bf16_t* Wkt    = (bf16_t*)(ws + SZ_XB + SZ_WT);
  bf16_t* Wvt    = (bf16_t*)(ws + SZ_XB + 2 * SZ_WT);
  bf16_t* Qb     = (bf16_t*)(ws + SZ_XB + 3 * SZ_WT);
  bf16_t* Kb     = (bf16_t*)(ws + SZ_XB + 3 * SZ_WT + SZ_H);
  bf16_t* Vtb    = (bf16_t*)(ws + SZ_XB + 3 * SZ_WT + 2 * SZ_H);
  bf16_t* AO     = (bf16_t*)(ws + SZ_XB + 3 * SZ_WT + 3 * SZ_H);
  float*  colsum = (float*)(ws + SZ_XB + 3 * SZ_WT + 4 * SZ_H);
  float*  yb     = (float*)(ws + SZ_XB + 3 * SZ_WT + 4 * SZ_H + (size_t)8192 * 4);

  hipMemsetAsync(colsum, 0, 8192 * sizeof(float), stream);

  k_cvt_x<<<4096, 256, 0, stream>>>(x, xb, 8192 * 1024 / 8);
  k_wT<<<4096, 256, 0, stream>>>(Wq, Wqt);
  k_wT<<<4096, 256, 0, stream>>>(Wk, Wkt);
  k_wT<<<4096, 256, 0, stream>>>(Wv, Wvt);
  k_qkv<<<dim3(16, 128, 3), 128, 0, stream>>>(xb, Wqt, Wkt, Wvt, bq, bk, bv, Qb, Kb, Vtb);
  k_attn<<<dim3(32, 64), 128, 0, stream>>>(Qb, Kb, Vtb, AO, colsum);
  k_pool<<<dim3(4, 4), 256, 0, stream>>>(colsum, AO, yb);
  k_oproj<<<dim3(4, 4), 256, 0, stream>>>(yb, Wo, bo, out);
}
